// ShuffleNetV2_50354196578344
// MI455X (gfx1250) — 8-bit matrix operand order repaired and checked by reading, compiled, not run on hardware
//
#include <hip/hip_runtime.h>
#include <cstdint>
#include <cstddef>


typedef int v8i __attribute__((ext_vector_type(8)));

__device__ __forceinline__ float fq_code(float x, float inv) {
    float r = rintf(x * inv);
    return fminf(fmaxf(r, -127.0f), 127.0f);
}

__global__ void k_quant_act(const float* __restrict__ in, int8_t* __restrict__ out,
                            int P, int Cin, int Cstride, int Coff, int Kpad, float inv) {
    int kq4 = Kpad >> 2;
    int t = blockIdx.x * blockDim.x + threadIdx.x;
    if (t >= P * kq4) return;
    int p  = t / kq4;
    int k0 = (t - p * kq4) * 4;
    const float* row = in + (size_t)p * Cstride + Coff;
    int pk = 0;
#pragma unroll
    for (int i = 0; i < 4; ++i) {
        int k = k0 + i;
        int q = 0;
        if (k < Cin) q = (int)fq_code(row[k], inv);
        pk |= (q & 0xff) << (8 * i);
    }
    ((int*)out)[(size_t)p * kq4 + (k0 >> 2)] = pk;
}

__global__ void k_quant_wgt(const float* __restrict__ w, int8_t* __restrict__ out,
                            int M, int K, int Mpad, int Kpad, float inv) {
    int kq4 = Kpad >> 2;
    int t = blockIdx.x * blockDim.x + threadIdx.x;
    if (t >= Mpad * kq4) return;
    int r  = t / kq4;
    int k0 = (t - r * kq4) * 4;
    int pk = 0;
#pragma unroll
    for (int i = 0; i < 4; ++i) {
        int k = k0 + i;
        int q = 0;
        if (r < M && k < K) q = (int)fq_code(w[(size_t)r * K + k], inv);
        pk |= (q & 0xff) << (8 * i);
    }
    ((int*)out)[t] = pk;
}

__device__ __forceinline__ v8i load_a_frag(const int8_t* __restrict__ arow, int k0) {
    const int* ap = (const int*)(arow + k0);
    v8i a;
    a[0] = ap[0];  a[1] = ap[1];  a[2] = ap[2];  a[3] = ap[3];
    a[4] = ap[8];  a[5] = ap[9];  a[6] = ap[10]; a[7] = ap[11];
    return a;
}
__device__ __forceinline__ v8i load_b_frag(const int8_t* __restrict__ bcol, int k0) {
    const int* bp = (const int*)(bcol + k0);
    v8i b;
    b[0] = bp[0];  b[1] = bp[1];  b[2] = bp[2];  b[3] = bp[3];
    b[4] = bp[8];  b[5] = bp[9];  b[6] = bp[10]; b[7] = bp[11];
    return b;
}

#define WMMA_I8(A, B, C) __builtin_amdgcn_wmma_i32_16x16x64_iu8(true, A, true, B, C, false, false)

__global__ void __launch_bounds__(256)
k_gemm_i8(const int8_t* __restrict__ Aq, const int8_t* __restrict__ Bq,
          float* __restrict__ out,
          int Mt2, int Nt2, int M, int Kpad, float scale,
          const float* __restrict__ bng, const float* __restrict__ bnb,
          const float* __restrict__ bnm, const float* __restrict__ bnv,
          const float* __restrict__ bias, int relu,
          int outCstride, int outBase, int outMul) {
    int wave = blockIdx.x * (blockDim.x >> 5) + (threadIdx.x >> 5);
    if (wave >= Mt2 * Nt2) return;
    int tm = wave / Nt2;
    int tn = wave - tm * Nt2;
    int lane = threadIdx.x & 31;
    int hf   = lane >> 4;
    int l15  = lane & 15;

    const int8_t* arow0 = Aq + (size_t)(tm * 32 + l15)      * Kpad + hf * 16;
    const int8_t* arow1 = Aq + (size_t)(tm * 32 + 16 + l15) * Kpad + hf * 16;
    const int8_t* bcol0 = Bq + (size_t)(tn * 32 + l15)      * Kpad + hf * 16;
    const int8_t* bcol1 = Bq + (size_t)(tn * 32 + 16 + l15) * Kpad + hf * 16;

    v8i acc00 = {0,0,0,0,0,0,0,0}, acc01 = acc00, acc10 = acc00, acc11 = acc00;

    v8i a0 = load_a_frag(arow0, 0), a1 = load_a_frag(arow1, 0);
    v8i b0 = load_b_frag(bcol0, 0), b1 = load_b_frag(bcol1, 0);
    for (int k0 = 0; k0 + 64 < Kpad; k0 += 64) {
        v8i na0 = load_a_frag(arow0, k0 + 64), na1 = load_a_frag(arow1, k0 + 64);
        v8i nb0 = load_b_frag(bcol0, k0 + 64), nb1 = load_b_frag(bcol1, k0 + 64);
        acc00 = WMMA_I8(a0, b0, acc00);
        acc01 = WMMA_I8(a0, b1, acc01);
        acc10 = WMMA_I8(a1, b0, acc10);
        acc11 = WMMA_I8(a1, b1, acc11);
        a0 = na0; a1 = na1; b0 = nb0; b1 = nb1;
    }
    acc00 = WMMA_I8(a0, b0, acc00);
    acc01 = WMMA_I8(a0, b1, acc01);
    acc10 = WMMA_I8(a1, b0, acc10);
    acc11 = WMMA_I8(a1, b1, acc11);

    int col0 = tn * 32 + l15;
    int col1 = col0 + 16;
#pragma unroll
    for (int r = 0; r < 8; ++r) {
        int m0 = tm * 32 + r + hf * 8;
        int m1 = m0 + 16;
#pragma unroll
        for (int q = 0; q < 4; ++q) {
            int mrow = (q & 2) ? m1 : m0;
            int col  = (q & 1) ? col1 : col0;
            int av   = (q == 0) ? acc00[r] : (q == 1) ? acc01[r]
                     : (q == 2) ? acc10[r] : acc11[r];
            if (mrow < M) {
                float f = (float)av * scale;
                if (bng) {
                    float sc = bng[mrow] * rsqrtf(bnv[mrow] + 1e-5f);
                    f = (f - bnm[mrow]) * sc + bnb[mrow];
                }
                if (bias) f += bias[mrow];
                if (relu) f = fmaxf(f, 0.0f);
                out[(size_t)col * outCstride + outBase + mrow * outMul] = f;
            }
        }
    }
}

__global__ void k_dwconv3(const float* __restrict__ in, const float* __restrict__ w,
                          float* __restrict__ out,
                          int Nimg, int H, int W, int C, int stride,
                          float invA, float sA, float invW, float sW,
                          const float* __restrict__ g, const float* __restrict__ b,
                          const float* __restrict__ m, const float* __restrict__ v,
                          int relu, int Ho, int Wo) {
    long long t = (long long)blockIdx.x * blockDim.x + threadIdx.x;
    long long tot = (long long)Nimg * Ho * Wo * C;
    if (t >= tot) return;
    int c  = (int)(t % C);
    long long rest = t / C;
    int wo = (int)(rest % Wo); rest /= Wo;
    int ho = (int)(rest % Ho);
    int n  = (int)(rest / Ho);
    float acc = 0.0f;
#pragma unroll
    for (int kh = 0; kh < 3; ++kh) {
#pragma unroll
        for (int kw = 0; kw < 3; ++kw) {
            int hi = ho * stride + kh - 1;
            int wi = wo * stride + kw - 1;
            if (hi >= 0 && hi < H && wi >= 0 && wi < W) {
                float xv = in[((size_t)(n * H + hi) * W + wi) * C + c];
                float xq = fq_code(xv, invA) * sA;
                float wq = fq_code(w[c * 9 + kh * 3 + kw], invW) * sW;
                acc += xq * wq;
            }
        }
    }
    float sc = g[c] * rsqrtf(v[c] + 1e-5f);
    float f = (acc - m[c]) * sc + b[c];
    if (relu) f = fmaxf(f, 0.0f);
    out[((size_t)(n * Ho + ho) * Wo + wo) * C + c] = f;
}

__global__ void k_preconv(const float* __restrict__ x, const float* __restrict__ w,
                          float* __restrict__ out,
                          float invA, float sA, float invW, float sW,
                          const float* __restrict__ g, const float* __restrict__ b,
                          const float* __restrict__ m, const float* __restrict__ v) {
    long long t = (long long)blockIdx.x * blockDim.x + threadIdx.x;
    long long tot = 512LL * 32 * 32 * 24;
    if (t >= tot) return;
    int co = (int)(t % 24);
    long long rest = t / 24;
    int ww = (int)(rest % 32); rest /= 32;
    int h  = (int)(rest % 32);
    int n  = (int)(rest / 32);
    float acc = 0.0f;
    for (int ci = 0; ci < 3; ++ci)
#pragma unroll
        for (int kh = 0; kh < 3; ++kh)
#pragma unroll
            for (int kw = 0; kw < 3; ++kw) {
                int hi = h + kh - 1, wi = ww + kw - 1;
                if (hi >= 0 && hi < 32 && wi >= 0 && wi < 32) {
                    float xv = x[((size_t)(n * 3 + ci) * 32 + hi) * 32 + wi];
                    float xq = fq_code(xv, invA) * sA;
                    float wq = fq_code(w[((co * 3 + ci) * 3 + kh) * 3 + kw], invW) * sW;
                    acc += xq * wq;
                }
            }
    float sc = g[co] * rsqrtf(v[co] + 1e-5f);
    out[((size_t)(n * 32 + h) * 32 + ww) * 24 + co] = (acc - m[co]) * sc + b[co];
}

__global__ void k_copy_ch(const float* __restrict__ in, float* __restrict__ out,
                          long long P, int c, int inCs, int inOff,
                          int outCs, int outBase, int outMul) {
    long long t = (long long)blockIdx.x * blockDim.x + threadIdx.x;
    if (t >= P * c) return;
    int i = (int)(t % c);
    long long p = t / c;
    out[(size_t)p * outCs + outBase + i * outMul] = in[(size_t)p * inCs + inOff + i];
}

__global__ void k_avgpool(const float* __restrict__ in, float* __restrict__ out,
                          int Nimg, int HW, int C) {
    int t = blockIdx.x * blockDim.x + threadIdx.x;
    if (t >= Nimg * C) return;
    int c = t % C, n = t / C;
    float s = 0.0f;
    for (int p = 0; p < HW; ++p) s += in[((size_t)n * HW + p) * C + c];
    out[t] = s / (float)HW;
}

static const float KA_RAW[56] = {
    2.7537312507629395f, 5.234059810638428f, 4.2632646560668945f, 9.554951667785645f,
    5.234059810638428f, 11.346541404724121f, 7.264077186584473f, 6.350336074829102f,
    9.16678237915039f, 5.25019645690918f, 3.478346347808838f, 5.273358345031738f,
    6.550426959991455f, 6.841591835021973f, 6.551841735839844f, 8.429676055908203f,
    3.6293373107910156f, 6.033638000488281f, 8.429676055908203f, 8.35006332397461f,
    3.1058642864227295f, 2.4753012657165527f, 4.7964186668396f, 3.1863455772399902f,
    2.5042595863342285f, 4.836341857910156f, 3.550802707672119f, 2.6628284454345703f,
    4.511188983917236f, 3.9728715419769287f, 2.248741626739502f, 4.604737758636475f,
    2.9986534118652344f, 2.4434714317321777f, 5.667801856994629f, 3.9212722778320312f,
    2.7176830768585205f, 5.555120944976807f, 3.8605222702026367f, 2.682180404663086f,
    5.483858585357666f, 5.311583518981934f, 2.37758731842041f, 7.243487358093262f,
    5.311583518981934f, 6.235666275024414f, 2.933143377304077f, 2.8617892265319824f,
    6.526449680328369f, 3.827115774154663f, 2.4493494033813477f, 6.2418341636657715f,
    3.028607130050659f, 2.5422754287719727f, 7.466217994689941f, 3.8511102199554443f};
static const float KW_RAW[56] = {
    1.3029288053512573f, 0.8977667689323425f, 1.2451014518737793f, 1.0910767316818237f,
    1.0148509740829468f, 1.4486286640167236f, 0.7227526307106018f, 0.9099447131156921f,
    0.8158362507820129f, 0.5097854733467102f, 0.9196422696113586f, 0.871035635471344f,
    0.9180927276611328f, 0.6798731684684753f, 0.9902427792549133f, 0.4500037431716919f,
    0.336839497089386f, 0.6030286550521851f, 0.47783952951431274f, 0.7545949220657349f,
    0.2900848984718323f, 0.6755377054214478f, 0.49563902616500854f, 0.3502323627471924f,
    0.4965823292732239f, 0.5626934766769409f, 0.4495861530303955f, 0.5228708386421204f,
    0.706261932849884f, 0.4907167851924896f, 0.48858827352523804f, 0.49457845091819763f,
    0.3859078288078308f, 0.4841776490211487f, 0.4749361276626587f, 0.625221848487854f,
    0.39950987696647644f, 0.45098355412483215f, 0.6184139847755432f, 0.3726619482040405f,
    0.42748335003852844f, 0.4202514886856079f, 0.27142152190208435f, 0.36484283208847046f,
    0.47397205233573914f, 0.5499554872512817f, 0.2817867696285248f, 0.39557644724845886f,
    0.4021666347980499f, 0.3414369523525238f, 0.37255141139030457f, 0.36935344338417053f,
    0.41747722029685974f, 0.29923439025878906f, 0.34784701466560364f, 0.2244890332221985f};

static inline float KAv(int i) { return (i < 56 ? KA_RAW[i] : 1.0f) / 15.0f; }
static inline float KWv(int i) { return (i < 56 ? KW_RAW[i] : 1.0f) / 15.0f; }
static inline float QINV(float K) { return (127.0f / 15.0f) / K; }
static inline float QSC(float K)  { return K * (15.0f / 127.0f); }
static inline int rup(int x, int a) { return (x + a - 1) / a * a; }
static inline int cdivl(long long a, long long b) { return (int)((a + b - 1) / b); }

extern "C" void kernel_launch(void* const* d_in, const int* in_sizes, int n_in,
                              void* d_out, int out_size, void* d_ws, size_t ws_size,
                              hipStream_t stream) {
    (void)in_sizes; (void)n_in; (void)out_size; (void)ws_size;
    auto F = [&](int i) { return (const float*)d_in[i]; };

    char* wsp = (char*)d_ws;
    auto carve = [&](size_t bytes) -> void* {
        void* p = (void*)wsp;
        wsp += (bytes + 255) / 256 * 256;
        return p;
    };
    float*  B0   = (float*)carve(64ull << 20);
    float*  B1   = (float*)carve(64ull << 20);
    float*  R0   = (float*)carve(64ull << 20);
    float*  R1   = (float*)carve(64ull << 20);
    int8_t* QA   = (int8_t*)carve(48ull << 20);
    int8_t* QW   = (int8_t*)carve(4ull << 20);
    float*  POOL = (float*)carve(4ull << 20);

    const int L_c5bn = 0, L_c5w = 4, L_fcb = 5, L_fcw = 6, L_prebn = 7, L_prew = 11;
    const int LX = 282;

    auto qact = [&](const float* in, long long P, int Cin, int Cs, int Coff,
                    int ka, int Kp) {
        long long t = P * (Kp / 4);
        k_quant_act<<<cdivl(t, 256), 256, 0, stream>>>(in, QA, (int)P, Cin, Cs, Coff,
                                                       Kp, QINV(KAv(ka)));
    };
    auto conv1x1 = [&](const float* in, long long P, int Cin, int Cs, int Coff,
                       int wleaf, int Cout, int kidx, float* out, int outCs,
                       int outBase, int outMul, int bnleaf, int relu,
                       const float* bias) {
        int Kp = rup(Cin, 64), Mp = rup(Cout, 32);
        qact(in, P, Cin, Cs, Coff, kidx, Kp);
        long long tw = (long long)Mp * (Kp / 4);
        k_quant_wgt<<<cdivl(tw, 256), 256, 0, stream>>>(F(wleaf), QW, Cout, Cin, Mp,
                                                        Kp, QINV(KWv(kidx)));
        int Mt2 = Mp / 32, Nt2 = (int)(P / 32);
        long long waves = (long long)Mt2 * Nt2;
        k_gemm_i8<<<cdivl(waves, 8), 256, 0, stream>>>(
            QW, QA, out, Mt2, Nt2, Cout, Kp, QSC(KAv(kidx)) * QSC(KWv(kidx)),
            bnleaf >= 0 ? F(bnleaf + 1) : nullptr,
            bnleaf >= 0 ? F(bnleaf + 0) : nullptr,
            bnleaf >= 0 ? F(bnleaf + 2) : nullptr,
            bnleaf >= 0 ? F(bnleaf + 3) : nullptr,
            bias, relu, outCs, outBase, outMul);
    };
    auto dwconv = [&](const float* in, int wleaf, float* out, int H, int C,
                      int stride, int kidx, int bnleaf) {
        int Ho = (stride == 2) ? H / 2 : H;
        long long tot = 512LL * Ho * Ho * C;
        k_dwconv3<<<cdivl(tot, 256), 256, 0, stream>>>(
            in, F(wleaf), out, 512, H, H, C, stride,
            QINV(KAv(kidx)), QSC(KAv(kidx)), QINV(KWv(kidx)), QSC(KWv(kidx)),
            F(bnleaf + 1), F(bnleaf + 0), F(bnleaf + 2), F(bnleaf + 3), 0, Ho, Ho);
    };

    {
        long long tot = 512LL * 32 * 32 * 24;
        k_preconv<<<cdivl(tot, 256), 256, 0, stream>>>(
            F(LX), F(L_prew), B0,
            QINV(KAv(0)), QSC(KAv(0)), QINV(KWv(0)), QSC(KWv(0)),
            F(L_prebn + 1), F(L_prebn + 0), F(L_prebn + 2), F(L_prebn + 3));
    }

    struct SS { int cin, cout, nunits, kbase; };
    const SS stages[3] = {{24, 116, 4, 1}, {116, 232, 8, 14}, {232, 464, 4, 40}};
    int leaf = 12;
    int H = 32;
    long long P = 512LL * 32 * 32;
    float* cur = B0;
    float* alt = B1;

    for (int s = 0; s < 3; ++s) {
        int cin = stages[s].cin, cout = stages[s].cout, half = cout / 2;
        int kb = stages[s].kbase;
        int L = leaf; leaf += 25;
        long long Pd = P / 4;
        int Hd = H / 2;
        conv1x1(cur, P, cin, cin, 0, L + 0, cin, kb, R0, cin, 0, 1, L + 3, 1, nullptr);
        dwconv(R0, L + 1, R1, H, cin, 2, kb + 1, L + 7);
        conv1x1(R1, Pd, cin, cin, 0, L + 2, half, kb + 2, alt, cout, 1, 2, L + 11, 1, nullptr);
        dwconv(cur, L + 15, R0, H, cin, 2, kb + 3, L + 17);
        conv1x1(R0, Pd, cin, cin, 0, L + 16, half, kb + 4, alt, cout, 0, 2, L + 21, 1, nullptr);
        { float* t = cur; cur = alt; alt = t; }
        P = Pd; H = Hd;

        int k = kb + 5;
        for (int u = 1; u < stages[s].nunits; ++u, k += 3) {
            int c = cout / 2;
            int Ln = leaf; leaf += 15;
            conv1x1(cur, P, c, cout, c, Ln + 0, c, k, R0, c, 0, 1, Ln + 3, 1, nullptr);
            dwconv(R0, Ln + 1, R1, H, c, 1, k + 1, Ln + 7);
            conv1x1(R1, P, c, c, 0, Ln + 2, c, k + 2, alt, cout, 1, 2, Ln + 11, 1, nullptr);
            k_copy_ch<<<cdivl(P * c, 256), 256, 0, stream>>>(cur, alt, P, c, cout, 0,
                                                             cout, 0, 2);
            { float* t = cur; cur = alt; alt = t; }
        }
    }

    conv1x1(cur, P, 464, 464, 0, L_c5w, 1024, 55, alt, 1024, 0, 1, L_c5bn, 1, nullptr);

    k_avgpool<<<cdivl(512LL * 1024, 256), 256, 0, stream>>>(alt, POOL, 512, 16, 1024);

    conv1x1(POOL, 512, 1024, 1024, 0, L_fcw, 100, 56, (float*)d_out, 100, 0, 1,
            -1, 0, F(L_fcb));
}
